// MS_Attention_RepConv_qkv_id_56573309224413
// MI455X (gfx1250) — matrix operand order repaired and checked by reading, not compiled, not run on hardware
//
#include <hip/hip_runtime.h>
#include <hip/hip_bf16.h>

typedef __attribute__((ext_vector_type(16))) _Float16 v16h;
typedef __attribute__((ext_vector_type(8)))  _Float16 v8h;
typedef __attribute__((ext_vector_type(8)))  float    v8f;

#define C_ 256
#define N_ 4096
#define T_ 8

// ---- WMMA helpers (CDNA5 v_wmma_f32_16x16x32_f16, wave32 layouts) ----------
// A operand as packed here (16x32 f16, same k as B): lane L holds row m=L%16; elements 0..15 = K (b..b+15),
// where b = (L>=16)*16.
__device__ __forceinline__ v16h make_a(const _Float16* __restrict__ row_k0, int bsel) {
    v8h lo = *(const v8h*)(row_k0 + 2 * bsel);
    v8h hi = *(const v8h*)(row_k0 + 2 * bsel + 8);
    v16h a;
#pragma unroll
    for (int i = 0; i < 8; ++i) { a[i] = lo[i]; a[8 + i] = hi[i]; }
    return a;
}

__device__ __forceinline__ v8f wmma_f16(v16h a, v16h b, v8f c) {
    return __builtin_amdgcn_wmma_f32_16x16x32_f16(false, a, false, b, (short)0, c, false, false);
}

// ---- prep: fp32->fp16 weights + fold BN params into per-channel affine -----
__global__ __launch_bounds__(256)
void k_prep(const float* __restrict__ w1, const float* __restrict__ pw,
            const float* __restrict__ bnp,
            _Float16* __restrict__ w1h, _Float16* __restrict__ pwh,
            float* __restrict__ aff /* [a1s|a1b|a2s|a2b] x 256 */) {
    int i = blockIdx.x * 256 + threadIdx.x;
    if (i < C_ * C_) { w1h[i] = (_Float16)w1[i]; pwh[i] = (_Float16)pw[i]; }
    if (blockIdx.x == 0) {
        int c = threadIdx.x;  // 256 threads == 256 channels
        // bn tensor layout: [3][4][C] with 0=w,1=b,2=m,3=var
        float s0 = bnp[(0*4+0)*C_+c] * rsqrtf(bnp[(0*4+3)*C_+c] + 1e-5f);
        float b0 = bnp[(0*4+1)*C_+c] - bnp[(0*4+2)*C_+c] * s0;
        float s1 = bnp[(1*4+0)*C_+c] * rsqrtf(bnp[(1*4+3)*C_+c] + 1e-5f);
        float b1 = bnp[(1*4+1)*C_+c] - bnp[(1*4+2)*C_+c] * s1;
        float s2 = bnp[(2*4+0)*C_+c] * rsqrtf(bnp[(2*4+3)*C_+c] + 1e-5f);
        float b2 = bnp[(2*4+1)*C_+c] - bnp[(2*4+2)*C_+c] * s2;
        aff[0*C_+c] = s0;            // scale after first 1x1
        aff[1*C_+c] = b0;            // bias (== pad value)
        aff[2*C_+c] = s1 * s2;       // fused double-BN scale after second 1x1
        aff[3*C_+c] = b1 * s2 + b2;  // fused double-BN bias
    }
}

// ---- LIF over t on raw f32 input x -> n-major f16 spikes [t][n][C] ---------
__global__ __launch_bounds__(256)
void k_lif_x(const float* __restrict__ x, _Float16* __restrict__ outT) {
    int gid = blockIdx.x * 256 + threadIdx.x;       // over C*N
    int c = gid / N_, n = gid % N_;
    float v = 0.f;
#pragma unroll
    for (int t = 0; t < T_; ++t) {
        float xt = x[(size_t)t * C_ * N_ + gid];
        v = 0.5f * (v + xt);                        // v += (x - v)/TAU, TAU=2
        float s = (v >= 1.0f) ? 1.0f : 0.0f;        // THR=1
        if (s > 0.f) v = 0.f;
        outT[(size_t)t * N_ * C_ + (size_t)n * C_ + c] = (_Float16)s;
    }
}

// ---- LIF over t on ch-major f16 input [t][c][n] -> spikes (layout flag) ----
template <bool NMAJOR>
__global__ __launch_bounds__(256)
void k_lif_spk(const _Float16* __restrict__ in, _Float16* __restrict__ out) {
    int gid = blockIdx.x * 256 + threadIdx.x;       // over C*N
    int c = gid / N_, n = gid % N_;
    float v = 0.f;
#pragma unroll
    for (int t = 0; t < T_; ++t) {
        float xt = (float)in[(size_t)t * C_ * N_ + gid];
        v = 0.5f * (v + xt);
        float s = (v >= 1.0f) ? 1.0f : 0.0f;
        if (s > 0.f) v = 0.f;
        size_t o = NMAJOR ? ((size_t)t * N_ * C_ + (size_t)n * C_ + c)
                          : ((size_t)t * C_ * N_ + (size_t)c * N_ + n);
        out[o] = (_Float16)s;
    }
}

// ---- 1x1-conv GEMM: Out[t][m][n] = aff(sum_k W[m][k] * In[t][n][k]) --------
// W: MxK row-major f16 (M=K=256). In: n-major [t][n][K]. Out: ch-major.
// Wave tile: 16(M) x 64(N), A reused across 4 WMMAs; K-loop 256/32 = 8.
template <bool F32OUT>
__global__ __launch_bounds__(256)
void k_gemm(const _Float16* __restrict__ W, const _Float16* __restrict__ In,
            void* __restrict__ Out, const float* __restrict__ scale,
            const float* __restrict__ bias) {
    const int K = C_;
    int lane = threadIdx.x & 31, wave = threadIdx.x >> 5;
    int t  = blockIdx.z;
    int m0 = blockIdx.y * 128 + wave * 16;
    int n0 = blockIdx.x * 64;
    const _Float16* inT = In + (size_t)t * N_ * K;

    int mA   = m0 + (lane & 15);
    int bsel = (lane >> 4) << 3;        // 0 or 8
    int ncol = lane & 15;
    int kbB  = (lane >> 4) << 4;        // 0 or 16

    v8f a0 = {0,0,0,0,0,0,0,0}, a1 = a0, a2 = a0, a3 = a0;
#pragma unroll
    for (int k0 = 0; k0 < K; k0 += 32) {
        v16h a = make_a(W + (size_t)mA * K + k0, bsel);
        const _Float16* bp = inT + (size_t)(n0 + ncol) * K + k0 + kbB;
        __builtin_prefetch((const void*)(bp + 32), 0, 1);
        v16h b0 = *(const v16h*)(bp);
        v16h b1 = *(const v16h*)(bp + 16 * K);
        v16h b2 = *(const v16h*)(bp + 32 * K);
        v16h b3 = *(const v16h*)(bp + 48 * K);
        a0 = wmma_f16(a, b0, a0);
        a1 = wmma_f16(a, b1, a1);
        a2 = wmma_f16(a, b2, a2);
        a3 = wmma_f16(a, b3, a3);
    }
    int rowAdd = (lane >> 4) << 3;      // 0 or 8
#pragma unroll
    for (int r = 0; r < 8; ++r) {
        int m = m0 + r + rowAdd;
        float sc = scale[m], bi = bias[m];
        size_t base = (size_t)t * C_ * N_ + (size_t)m * N_ + n0 + ncol;
        float y0 = a0[r] * sc + bi, y1 = a1[r] * sc + bi;
        float y2 = a2[r] * sc + bi, y3 = a3[r] * sc + bi;
        if (F32OUT) {
            float* fo = (float*)Out;
            fo[base] = y0; fo[base + 16] = y1; fo[base + 32] = y2; fo[base + 48] = y3;
        } else {
            _Float16* ho = (_Float16*)Out;
            ho[base] = (_Float16)y0; ho[base + 16] = (_Float16)y1;
            ho[base + 32] = (_Float16)y2; ho[base + 48] = (_Float16)y3;
        }
    }
}

// ---- 3x3 depthwise conv with per-channel pad value -------------------------
// in: ch-major [t][c][64][64] f16; out: n-major [t][n][C] f16
__global__ __launch_bounds__(256)
void k_dw(const _Float16* __restrict__ in, const float* __restrict__ dwW,
          const float* __restrict__ pv, _Float16* __restrict__ out) {
    int n = blockIdx.x * 256 + threadIdx.x;
    int c = blockIdx.y, t = blockIdx.z;
    int h = n >> 6, w = n & 63;
    const _Float16* ip = in + ((size_t)t * C_ + c) * (size_t)N_;
    float p = pv[c];
    float acc = 0.f;
#pragma unroll
    for (int i = 0; i < 3; ++i) {
#pragma unroll
        for (int j = 0; j < 3; ++j) {
            int hh = h - 1 + i, ww = w - 1 + j;
            float val = (hh >= 0 && hh < 64 && ww >= 0 && ww < 64)
                        ? (float)ip[hh * 64 + ww] : p;
            acc += dwW[c * 9 + i * 3 + j] * val;
        }
    }
    out[(size_t)t * N_ * C_ + (size_t)n * C_ + c] = (_Float16)acc;
}

// ---- kv = K^T V per (t,head): M=32,N=32,K=4096; spikes ch-major ------------
__global__ __launch_bounds__(128)
void k_kv(const _Float16* __restrict__ kS, const _Float16* __restrict__ vS,
          _Float16* __restrict__ kv) {
    int lane = threadIdx.x & 31, wave = threadIdx.x >> 5;
    int th = blockIdx.x, t = th >> 3, h = th & 7;
    int d0 = (wave & 1) * 16, e0 = (wave >> 1) * 16;
    const _Float16* kp = kS + ((size_t)t * C_ + h * 32) * (size_t)N_;
    const _Float16* vp = vS + ((size_t)t * C_ + h * 32) * (size_t)N_;
    int mA = d0 + (lane & 15), bsel = (lane >> 4) << 3;
    int ncol = e0 + (lane & 15), kbB = (lane >> 4) << 4;
    v8f acc = {0,0,0,0,0,0,0,0};
    for (int k0 = 0; k0 < N_; k0 += 32) {
        v16h a = make_a(kp + (size_t)mA * N_ + k0, bsel);
        v16h b = *(const v16h*)(vp + (size_t)ncol * N_ + k0 + kbB);
        acc = wmma_f16(a, b, acc);
    }
    int rowAdd = (lane >> 4) << 3;
#pragma unroll
    for (int r = 0; r < 8; ++r)
        kv[(size_t)th * 1024 + (d0 + r + rowAdd) * 32 + ncol] = (_Float16)acc[r];
}

// ---- o = Q * kv * scale per (t,head): M=4096,K=32,N=32; kv staged in LDS ---
__global__ __launch_bounds__(256)
void k_o(const _Float16* __restrict__ qS /*[t][n][C]*/,
         const _Float16* __restrict__ kv, _Float16* __restrict__ o /*[t][c][n]*/) {
    __shared__ _Float16 skv[1024];
    int th = blockIdx.y, t = th >> 3, h = th & 7;
    for (int i = threadIdx.x; i < 512; i += 256)
        ((float*)skv)[i] = ((const float*)(kv + (size_t)th * 1024))[i];
    __syncthreads();

    int lane = threadIdx.x & 31, wave = threadIdx.x >> 5;
    int m0 = blockIdx.x * 128 + wave * 16;          // n-row tile
    int mA = m0 + (lane & 15), bsel = (lane >> 4) << 3;
    const _Float16* qp = qS + ((size_t)t * N_ + mA) * C_ + h * 32;
    v16h a = make_a(qp, bsel);

    int ncol = lane & 15, kbB = (lane >> 4) << 4;
    v16h b0, b1;
#pragma unroll
    for (int i = 0; i < 16; ++i) {
        b0[i] = skv[(kbB + i) * 32 + ncol];
        b1[i] = skv[(kbB + i) * 32 + 16 + ncol];
    }
    v8f a0 = {0,0,0,0,0,0,0,0}, a1 = a0;
    a0 = wmma_f16(a, b0, a0);
    a1 = wmma_f16(a, b1, a1);

    const float s = 0.17677669529663687f;           // 32^-0.5
    int rowAdd = (lane >> 4) << 3;
#pragma unroll
    for (int r = 0; r < 8; ++r) {
        int n = m0 + r + rowAdd;
        o[((size_t)t * C_ + h * 32 + ncol) * N_ + n]      = (_Float16)(a0[r] * s);
        o[((size_t)t * C_ + h * 32 + 16 + ncol) * N_ + n] = (_Float16)(a1[r] * s);
    }
}

// ---------------------------------------------------------------------------
extern "C" void kernel_launch(void* const* d_in, const int* in_sizes, int n_in,
                              void* d_out, int out_size, void* d_ws, size_t ws_size,
                              hipStream_t stream) {
    (void)in_sizes; (void)n_in; (void)out_size; (void)ws_size;
    const size_t MB = 1ull << 20;
    char* ws = (char*)d_ws;

    _Float16* W1H = (_Float16*)(ws);                       // 4 x 64K f16
    _Float16* PWH = (_Float16*)(ws + 524288);              // 4 x 64K f16
    float*    AFF = (float*)   (ws + 1048576);             // 4 x 1024 f32
    _Float16* KV  = (_Float16*)(ws + 1048576 + 65536);     // 64 x 1024 f16
    char* big = ws + 2 * MB;
    _Float16* XST  = (_Float16*)(big + 0 * 16 * MB);       // n-major spikes / ch-major reuse
    _Float16* BUFA = (_Float16*)(big + 1 * 16 * MB);
    _Float16* BUFB = (_Float16*)(big + 2 * 16 * MB);
    _Float16* QS   = (_Float16*)(big + 3 * 16 * MB);
    _Float16* KS   = (_Float16*)(big + 4 * 16 * MB);
    _Float16* VS   = (_Float16*)(big + 5 * 16 * MB);

    // prep weights + BN folding, per branch (q,k,v,p at d_in[1+4b..4+4b])
    for (int br = 0; br < 4; ++br) {
        k_prep<<<256, 256, 0, stream>>>(
            (const float*)d_in[1 + 4 * br], (const float*)d_in[3 + 4 * br],
            (const float*)d_in[4 + 4 * br],
            W1H + (size_t)br * C_ * C_, PWH + (size_t)br * C_ * C_,
            AFF + (size_t)br * 1024);
    }

    // xs = lif(x) -> n-major f16 spikes
    k_lif_x<<<(C_ * N_) / 256, 256, 0, stream>>>((const float*)d_in[0], XST);

    dim3 ggrid(N_ / 64, C_ / 128, T_);
    dim3 dgrid(N_ / 256, C_, T_);
    _Float16* spkOut[3] = {QS, KS, VS};
    for (int br = 0; br < 3; ++br) {   // q, k, v branches
        const float* aff = AFF + (size_t)br * 1024;
        k_gemm<false><<<ggrid, 256, 0, stream>>>(
            W1H + (size_t)br * C_ * C_, XST, BUFA, aff + 0, aff + 256);
        k_dw<<<dgrid, 256, 0, stream>>>(
            BUFA, (const float*)d_in[2 + 4 * br], aff + 256, BUFB);
        k_gemm<false><<<ggrid, 256, 0, stream>>>(
            PWH + (size_t)br * C_ * C_, BUFB, BUFA, aff + 512, aff + 768);
        if (br == 0) k_lif_spk<true ><<<(C_ * N_) / 256, 256, 0, stream>>>(BUFA, spkOut[br]);
        else         k_lif_spk<false><<<(C_ * N_) / 256, 256, 0, stream>>>(BUFA, spkOut[br]);
    }

    // attention: kv = K^T V ; o = Q kv * d^-1/2 (ch-major into BUFA)
    k_kv<<<T_ * 8, 128, 0, stream>>>(KS, VS, KV);
    k_o<<<dim3(N_ / 128, T_ * 8), 256, 0, stream>>>(QS, KV, BUFA);

    // o = lif(o) -> n-major spikes in BUFB, then p-branch to f32 output
    k_lif_spk<true><<<(C_ * N_) / 256, 256, 0, stream>>>(BUFA, BUFB);
    const float* affp = AFF + 3 * 1024;
    k_gemm<false><<<ggrid, 256, 0, stream>>>(W1H + 3 * (size_t)C_ * C_, BUFB, XST, affp + 0, affp + 256);
    k_dw<<<dgrid, 256, 0, stream>>>(XST, (const float*)d_in[14], affp + 256, BUFB);
    k_gemm<true><<<ggrid, 256, 0, stream>>>(PWH + 3 * (size_t)C_ * C_, BUFB, (float*)d_out, affp + 512, affp + 768);
}
